// MultiheadAttention_1632087573280
// MI455X (gfx1250) — matrix operand order repaired and checked by reading, not compiled, not run on hardware
//
#include <hip/hip_runtime.h>

// Multi-head attention forward (S=1024, B=8, E=1024, H=16, hd=64) for gfx1250.
// Compute-bound (~103 GFLOP, <10us of HBM traffic at 23.3 TB/s) -> all GEMMs
// run on v_wmma_f32_16x16x32_bf16 (8x the MACs/instr of the f32 16x16x4 WMMA),
// bf16 inputs with fp32 accumulation; softmax in fp32.
// K-chunks of 64 -> 8 WMMAs per barrier pair; bf16 LDS staging uses
// global_load_async_to_lds_b128 (ASYNCcnt) where available.

#define S_  1024
#define B_  8
#define E_  1024
#define H_  16
#define HD_ 64
#define SB_ (S_ * B_)
#define E3_ (3 * E_)
#define BH_ (B_ * H_)

typedef __attribute__((ext_vector_type(16))) __bf16 v16bf;
typedef __attribute__((ext_vector_type(8)))  float  v8f;

union BFrag {
  v16bf v;
  unsigned int u[8];
  uint4 q[2];
};

__device__ __forceinline__ v8f vzero8() {
  v8f z = {0.f, 0.f, 0.f, 0.f, 0.f, 0.f, 0.f, 0.f};
  return z;
}

__device__ __forceinline__ v8f wmma_bf16(v16bf a, v16bf b, v8f c) {
  // (neg_a, A, neg_b, B, c_mod, C, reuse_a, reuse_b)
  return __builtin_amdgcn_wmma_f32_16x16x32_bf16(false, a, false, b, (short)0, c,
                                                 false, false);
}

// ---- gfx1250 async global->LDS copy (16 bytes / lane), ASYNCcnt-tracked ----
#if defined(__has_builtin)
#if __has_builtin(__builtin_amdgcn_global_load_async_to_lds_b128)
#define HAVE_ASYNC_LDS 1
#endif
#endif

typedef int v4i_vs __attribute__((vector_size(4 * sizeof(int))));

__device__ __forceinline__ void async_cp16(const void* g, void* l) {
#ifdef HAVE_ASYNC_LDS
  __builtin_amdgcn_global_load_async_to_lds_b128(
      (__attribute__((address_space(1))) v4i_vs*)g,
      (__attribute__((address_space(3))) v4i_vs*)l, 0, 0);
#else
  *reinterpret_cast<uint4*>(l) = *reinterpret_cast<const uint4*>(g);
#endif
}

__device__ __forceinline__ void wait_async0() {
#if defined(__has_builtin) && __has_builtin(__builtin_amdgcn_s_wait_asynccnt)
  __builtin_amdgcn_s_wait_asynccnt(0);
#else
#ifdef HAVE_ASYNC_LDS
  asm volatile("s_wait_asynccnt 0x0" ::: "memory");
#endif
#endif
}

// A fragment: 16xK row-major bf16 tile. ISA 7.12.2: lane -> M = lane&15,
// wave halves hold k-offset +0/+8, element pairs p -> k = (p<4 ? 2p : 2p+8).
__device__ __forceinline__ v16bf load_a_bf16(const __bf16* base, int stride,
                                             int row0, int k0, int lane) {
  BFrag r;
  const __bf16* p = base + (row0 + (lane & 15)) * stride + k0 + ((lane >> 4) << 3);
#pragma unroll
  for (int pr = 0; pr < 8; ++pr) {
    const int k = (pr < 4) ? (2 * pr) : (2 * pr + 8);
    r.u[pr] = *reinterpret_cast<const unsigned int*>(p + k);
  }
  return r.v;
}

// Same A layout, fp32 source converted on the fly (used for softmax probs).
__device__ __forceinline__ v16bf load_a_f32(const float* base, int stride,
                                            int row0, int k0, int lane) {
  BFrag r;
  const float* p = base + (row0 + (lane & 15)) * stride + k0 + ((lane >> 4) << 3);
#pragma unroll
  for (int pr = 0; pr < 8; ++pr) {
    const int k = (pr < 4) ? (2 * pr) : (2 * pr + 8);
    float2 f = *reinterpret_cast<const float2*>(p + k);
    union { unsigned u; __bf16 h[2]; } t;
    t.h[0] = (__bf16)f.x;
    t.h[1] = (__bf16)f.y;
    r.u[pr] = t.u;
  }
  return r.v;
}

// B fragment: Kx16 tile stored n-major (Bmat[n][k], k contiguous, row stride
// multiple of 8 elems). ISA layout: lane -> N = lane&15, wave halves hold
// k-offset +0/+8, element e -> k = (e<8 ? e : e+8).  Row-major (outF, inF) weights and the
// head-transposed V buffer feed this directly -- no transposes needed.
__device__ __forceinline__ v16bf load_b_bf16(const __bf16* base, int stride,
                                             int n0, int k0, int lane) {
  BFrag r;
  const __bf16* p = base + (long long)(n0 + (lane & 15)) * stride
                         + k0 + ((lane >> 4) << 3);
  r.q[0] = *reinterpret_cast<const uint4*>(p);
  r.q[1] = *reinterpret_cast<const uint4*>(p + 16);
  return r.v;
}

// Convert 16 consecutive fp32 -> 16 bf16 and store as two uint4 into LDS.
__device__ __forceinline__ void cvt16_to_lds(const float* src, __bf16* dst) {
  union { __bf16 h[16]; uint4 u[2]; } pk;
#pragma unroll
  for (int i = 0; i < 4; ++i) {
    float4 f = reinterpret_cast<const float4*>(src)[i];
    pk.h[4 * i + 0] = (__bf16)f.x; pk.h[4 * i + 1] = (__bf16)f.y;
    pk.h[4 * i + 2] = (__bf16)f.z; pk.h[4 * i + 3] = (__bf16)f.w;
  }
  uint4* d = reinterpret_cast<uint4*>(dst);
  d[0] = pk.u[0];
  d[1] = pk.u[1];
}

// ---------------------------------------------------------------------------
// Kernel 1: qkv = query @ Win^T + bias; scatter into head-major q/k (bf16,
// [bh][s][d], q pre-scaled) and transposed v (bf16, [bh][d][s]).
// Workgroup: 256 thr = 8 waves, tile 64(M) x 128(N), K-chunks of 64.
// ---------------------------------------------------------------------------
__global__ __launch_bounds__(256) void gemm_qkv_kernel(
    const float* __restrict__ q, const float* __restrict__ w,
    const float* __restrict__ bias, __bf16* __restrict__ qh,
    __bf16* __restrict__ kh, __bf16* __restrict__ vt) {
  __shared__ __align__(16) __bf16 As[64 * 72];   // 64 x 64 (+8 pad)
  __shared__ __align__(16) __bf16 Bs[128 * 72];  // 128 x 64 (+8 pad)
  const int tid = threadIdx.x, lane = tid & 31, wid = tid >> 5;
  const int m0 = blockIdx.y * 64;
  const int n0 = blockIdx.x * 128;
  const int wm = (wid >> 2) * 32, wn = (wid & 3) * 32;

  v8f acc[2][2];
  acc[0][0] = vzero8(); acc[0][1] = vzero8();
  acc[1][0] = vzero8(); acc[1][1] = vzero8();

  const int ar = tid >> 2, ac = (tid & 3) * 16;  // A: 64 rows x 64 cols, 16/thr
  const int bn = tid >> 1, bc = (tid & 1) * 32;  // B: 128 rows x 64 cols, 32/thr
  const float* asrc = q + (long long)(m0 + ar) * E_ + ac;
  const float* bsrc = w + (long long)(n0 + bn) * E_ + bc;

  for (int k0 = 0; k0 < E_; k0 += 64) {
    cvt16_to_lds(asrc + k0, &As[ar * 72 + ac]);
    cvt16_to_lds(bsrc + k0, &Bs[bn * 72 + bc]);
    cvt16_to_lds(bsrc + k0 + 16, &Bs[bn * 72 + bc + 16]);
    if (k0 + 64 < E_) {  // prefetch next chunk (global_prefetch_b8)
      __builtin_prefetch(asrc + k0 + 64, 0, 0);
      __builtin_prefetch(bsrc + k0 + 64, 0, 0);
    }
    __syncthreads();
    v16bf a00 = load_a_bf16(As, 72, wm, 0, lane);
    v16bf a01 = load_a_bf16(As, 72, wm, 32, lane);
    v16bf a10 = load_a_bf16(As, 72, wm + 16, 0, lane);
    v16bf a11 = load_a_bf16(As, 72, wm + 16, 32, lane);
    v16bf b00 = load_b_bf16(Bs, 72, wn, 0, lane);
    v16bf b01 = load_b_bf16(Bs, 72, wn, 32, lane);
    v16bf b10 = load_b_bf16(Bs, 72, wn + 16, 0, lane);
    v16bf b11 = load_b_bf16(Bs, 72, wn + 16, 32, lane);
    acc[0][0] = wmma_bf16(a00, b00, acc[0][0]);
    acc[0][0] = wmma_bf16(a01, b01, acc[0][0]);
    acc[0][1] = wmma_bf16(a00, b10, acc[0][1]);
    acc[0][1] = wmma_bf16(a01, b11, acc[0][1]);
    acc[1][0] = wmma_bf16(a10, b00, acc[1][0]);
    acc[1][0] = wmma_bf16(a11, b01, acc[1][0]);
    acc[1][1] = wmma_bf16(a10, b10, acc[1][1]);
    acc[1][1] = wmma_bf16(a11, b11, acc[1][1]);
    __syncthreads();
  }

  const float scaling = 0.125f;  // hd^-0.5
#pragma unroll
  for (int sm = 0; sm < 2; ++sm) {
#pragma unroll
    for (int sn = 0; sn < 2; ++sn) {
      const int f = n0 + wn + sn * 16 + (lane & 15);
      const int mb = m0 + wm + sm * 16 + ((lane >> 4) << 3);
      const float bv = bias[f];
#pragma unroll
      for (int v = 0; v < 8; ++v) {
        const int m = mb + v;
        const int s = m >> 3, b = m & 7;  // m = s*B + b
        const float val = acc[sm][sn][v] + bv;
        if (f < E_) {
          const int bh = b * H_ + (f >> 6), d = f & 63;
          qh[(long long)(bh * S_ + s) * HD_ + d] = (__bf16)(val * scaling);
        } else if (f < 2 * E_) {
          const int fk = f - E_;
          const int bh = b * H_ + (fk >> 6), d = fk & 63;
          kh[(long long)(bh * S_ + s) * HD_ + d] = (__bf16)val;
        } else {
          const int fv = f - 2 * E_;
          const int bh = b * H_ + (fv >> 6), d = fv & 63;
          vt[(long long)(bh * HD_ + d) * S_ + s] = (__bf16)val;
        }
      }
    }
  }
}

// ---------------------------------------------------------------------------
// Kernel 2: attention. One workgroup per (batch, 16-row q tile); loops heads.
// Scores row-block (16x1024 fp32) lives in LDS; softmax per-wave (2 rows each)
// with wave32 shuffles; averaged probs accumulate in registers.
// ---------------------------------------------------------------------------
__global__ __launch_bounds__(256) void attn_kernel(
    const __bf16* __restrict__ qh, const __bf16* __restrict__ kh,
    const __bf16* __restrict__ vt, __bf16* __restrict__ ctx,
    float* __restrict__ avg_out) {
  __shared__ __align__(16) float sc[16 * 1028];  // padded stride vs banks
  __shared__ __align__(16) __bf16 qt[16 * 72];   // stride 72 -> 16B rows
  const int tid = threadIdx.x, lane = tid & 31, wid = tid >> 5;
  const int b = blockIdx.x >> 6;
  const int s0 = (blockIdx.x & 63) << 4;

  float avg0[32], avg1[32];
#pragma unroll
  for (int i = 0; i < 32; ++i) { avg0[i] = 0.f; avg1[i] = 0.f; }

  for (int h = 0; h < H_; ++h) {
    const int bh = b * H_ + h;
    const __bf16* qhp = qh + (long long)bh * S_ * HD_;
    const __bf16* khp = kh + (long long)bh * S_ * HD_;
    const __bf16* vtp = vt + (long long)bh * HD_ * S_;
    __syncthreads();     // previous head done reading sc/qt
    if (tid < 128) {     // stage 16x64 Q tile: async global->LDS, 16B/lane
      const int idx = tid * 8, r = idx >> 6, d = idx & 63;
      async_cp16(qhp + (long long)(s0 + r) * HD_ + d, &qt[r * 72 + d]);
      wait_async0();
    }
    __syncthreads();
    const v16bf a0 = load_a_bf16(qt, 72, 0, 0, lane);
    const v16bf a1 = load_a_bf16(qt, 72, 0, 32, lane);

    // Phase 1: scores = (Q*scale) @ K^T, 64 key tiles over 8 waves
    for (int j = wid; j < 64; j += 8) {
      v8f c = vzero8();
      c = wmma_bf16(a0, load_b_bf16(khp, HD_, j * 16, 0, lane), c);
      c = wmma_bf16(a1, load_b_bf16(khp, HD_, j * 16, 32, lane), c);
      const int n = j * 16 + (lane & 15);
      const int mb = (lane >> 4) << 3;
#pragma unroll
      for (int v = 0; v < 8; ++v) sc[(mb + v) * 1028 + n] = c[v];
    }
    __syncthreads();

    // Phase 2: softmax, wave handles rows 2*wid and 2*wid+1
#pragma unroll
    for (int rr = 0; rr < 2; ++rr) {
      float* srow = &sc[(wid * 2 + rr) * 1028];
      float* areg = rr ? avg1 : avg0;
      float mx = -3.0e38f;
      for (int ii = 0; ii < 32; ++ii) mx = fmaxf(mx, srow[lane + 32 * ii]);
#pragma unroll
      for (int o = 16; o > 0; o >>= 1) mx = fmaxf(mx, __shfl_xor(mx, o, 32));
      float sum = 0.f;
      for (int ii = 0; ii < 32; ++ii) {
        const float e = __expf(srow[lane + 32 * ii] - mx);
        srow[lane + 32 * ii] = e;
        sum += e;
      }
#pragma unroll
      for (int o = 16; o > 0; o >>= 1) sum += __shfl_xor(sum, o, 32);
      const float inv = 1.0f / sum;
      for (int ii = 0; ii < 32; ++ii) {
        const float p = srow[lane + 32 * ii] * inv;
        srow[lane + 32 * ii] = p;
        areg[ii] += p * (1.0f / H_);
      }
    }
    __syncthreads();

    // Phase 3: attn = P @ V (16x64); waves 0..3 take one 16-col tile each
    if (wid < 4) {
      const int d0 = wid * 16;
      v8f c = vzero8();
      for (int k0 = 0; k0 < S_; k0 += 32) {
        v16bf a = load_a_f32(sc, 1028, 0, k0, lane);
        v16bf bb = load_b_bf16(vtp, S_, d0, k0, lane);
        c = wmma_bf16(a, bb, c);
      }
      const int d = d0 + (lane & 15);
      const int mb = (lane >> 4) << 3;
#pragma unroll
      for (int v = 0; v < 8; ++v) {
        const int s = s0 + mb + v;
        ctx[(long long)(s * B_ + b) * E_ + h * HD_ + d] = (__bf16)c[v];
      }
    }
  }

  // Averaged attention weights (per-wave private rows; no barrier needed)
  {
    const long long base = (long long)b * S_ * S_;
    const int r0 = s0 + wid * 2;
    for (int ii = 0; ii < 32; ++ii) {
      avg_out[base + (long long)(r0 + 0) * S_ + lane + 32 * ii] = avg0[ii];
      avg_out[base + (long long)(r0 + 1) * S_ + lane + 32 * ii] = avg1[ii];
    }
  }
}

// ---------------------------------------------------------------------------
// Kernel 3: out = ctx @ Wout^T + bias (fp32 result to d_out)
// ---------------------------------------------------------------------------
__global__ __launch_bounds__(256) void gemm_out_kernel(
    const __bf16* __restrict__ ctx, const float* __restrict__ w,
    const float* __restrict__ bias, float* __restrict__ out) {
  __shared__ __align__(16) __bf16 As[64 * 72];
  __shared__ __align__(16) __bf16 Bs[128 * 72];
  const int tid = threadIdx.x, lane = tid & 31, wid = tid >> 5;
  const int m0 = blockIdx.y * 64;
  const int n0 = blockIdx.x * 128;
  const int wm = (wid >> 2) * 32, wn = (wid & 3) * 32;

  v8f acc[2][2];
  acc[0][0] = vzero8(); acc[0][1] = vzero8();
  acc[1][0] = vzero8(); acc[1][1] = vzero8();

  const int ar = tid >> 2, ac = (tid & 3) * 16;
  const int bn = tid >> 1, bc = (tid & 1) * 32;
  const __bf16* asrc = ctx + (long long)(m0 + ar) * E_ + ac;
  const float* bsrc = w + (long long)(n0 + bn) * E_ + bc;

  for (int k0 = 0; k0 < E_; k0 += 64) {
    // A tile is already bf16: async global->LDS copy (2 x 16B per thread)
    async_cp16(asrc + k0, &As[ar * 72 + ac]);
    async_cp16(asrc + k0 + 8, &As[ar * 72 + ac + 8]);
    // B tile: fp32 -> bf16 conversion while staging
    cvt16_to_lds(bsrc + k0, &Bs[bn * 72 + bc]);
    cvt16_to_lds(bsrc + k0 + 16, &Bs[bn * 72 + bc + 16]);
    if (k0 + 64 < E_) {
      __builtin_prefetch(asrc + k0 + 64, 0, 0);
      __builtin_prefetch(bsrc + k0 + 64, 0, 0);
    }
    wait_async0();
    __syncthreads();
    v16bf a00 = load_a_bf16(As, 72, wm, 0, lane);
    v16bf a01 = load_a_bf16(As, 72, wm, 32, lane);
    v16bf a10 = load_a_bf16(As, 72, wm + 16, 0, lane);
    v16bf a11 = load_a_bf16(As, 72, wm + 16, 32, lane);
    v16bf b00 = load_b_bf16(Bs, 72, wn, 0, lane);
    v16bf b01 = load_b_bf16(Bs, 72, wn, 32, lane);
    v16bf b10 = load_b_bf16(Bs, 72, wn + 16, 0, lane);
    v16bf b11 = load_b_bf16(Bs, 72, wn + 16, 32, lane);
    acc[0][0] = wmma_bf16(a00, b00, acc[0][0]);
    acc[0][0] = wmma_bf16(a01, b01, acc[0][0]);
    acc[0][1] = wmma_bf16(a00, b10, acc[0][1]);
    acc[0][1] = wmma_bf16(a01, b11, acc[0][1]);
    acc[1][0] = wmma_bf16(a10, b00, acc[1][0]);
    acc[1][0] = wmma_bf16(a11, b01, acc[1][0]);
    acc[1][1] = wmma_bf16(a10, b10, acc[1][1]);
    acc[1][1] = wmma_bf16(a11, b11, acc[1][1]);
    __syncthreads();
  }

#pragma unroll
  for (int sm = 0; sm < 2; ++sm) {
#pragma unroll
    for (int sn = 0; sn < 2; ++sn) {
      const int f = n0 + wn + sn * 16 + (lane & 15);
      const int mb = m0 + wm + sm * 16 + ((lane >> 4) << 3);
      const float bv = bias[f];
#pragma unroll
      for (int v = 0; v < 8; ++v) {
        out[(long long)(mb + v) * E_ + f] = acc[sm][sn][v] + bv;
      }
    }
  }
}

extern "C" void kernel_launch(void* const* d_in, const int* in_sizes, int n_in,
                              void* d_out, int out_size, void* d_ws,
                              size_t ws_size, hipStream_t stream) {
  (void)in_sizes; (void)n_in; (void)out_size; (void)ws_size;
  const float* query = (const float*)d_in[0];
  const float* w_in  = (const float*)d_in[1];
  const float* b_in  = (const float*)d_in[2];
  const float* w_out = (const float*)d_in[3];
  const float* b_out = (const float*)d_in[4];

  float* out = (float*)d_out;                 // (S,B,E) flat
  float* avg = out + (size_t)SB_ * E_;        // (B,S,S) flat

  char* ws = (char*)d_ws;                     // 64 MB used total
  const size_t HBYTES = (size_t)BH_ * S_ * HD_ * 2;  // 16 MB each
  __bf16* qh  = (__bf16*)(ws);                // [bh][s][d], pre-scaled
  __bf16* kh  = (__bf16*)(ws + HBYTES);       // [bh][s][d]
  __bf16* vt  = (__bf16*)(ws + 2 * HBYTES);   // [bh][d][s]
  __bf16* ctx = (__bf16*)(ws + 3 * HBYTES);   // [(s*B+b)][e]

  gemm_qkv_kernel<<<dim3(E3_ / 128, SB_ / 64), 256, 0, stream>>>(
      query, w_in, b_in, qh, kh, vt);
  attn_kernel<<<dim3(B_ * (S_ / 16)), 256, 0, stream>>>(qh, kh, vt, ctx, avg);
  gemm_out_kernel<<<dim3(E_ / 128, SB_ / 64), 256, 0, stream>>>(
      ctx, w_out, b_out, out);
}
